// localfeaFromImage_44375602102906
// MI455X (gfx1250) — matrix operand order repaired and checked by reading, not compiled, not run on hardware
//
#include <hip/hip_runtime.h>
#include <hip/hip_bf16.h>
#include <math.h>

#define BATCH 128
#define KPTS  16
#define HW    224
#define PATCH 31

typedef __attribute__((ext_vector_type(16))) _Float16 v16h;
typedef __attribute__((ext_vector_type(8)))  float    v8f;

// ---------------- Kernel 1: relu + global 7x7 mean ----------------
// 8 lanes cooperate on one (b,c) row of 49 elements; 32 rows per 256-thread block.
__global__ void fmean_kernel(const float* __restrict__ features,
                             float* __restrict__ outmean) {
  const int tid = threadIdx.x;
  const int row = blockIdx.x * 32 + (tid >> 3);      // b*2208 + c
  const int j = tid & 7;
  if (row >= BATCH * 2208) return;                   // grid is exact; never taken
  const float* p = features + (size_t)row * 49;
  float s = 0.f;
  for (int i = j; i < 49; i += 8) s += fmaxf(p[i], 0.f);
  s += __shfl_xor(s, 1, 8);
  s += __shfl_xor(s, 2, 8);
  s += __shfl_xor(s, 4, 8);
  if (j == 0) outmean[row] = s * (1.f / 49.f);
}

// ---------------- Kernel 2: crop + conv1/bn/relu/maxpool + conv2/bn/relu/avgpool --
// One block per (b,k). Everything staged in LDS; bn folded to scale/offset.
__global__ void __launch_bounds__(128)
pconv_kernel(const float* __restrict__ x, const float* __restrict__ points,
             const float* __restrict__ c1w, const float* __restrict__ c1b,
             const float* __restrict__ s1, const float* __restrict__ bb1,
             const float* __restrict__ m1, const float* __restrict__ v1,
             const float* __restrict__ c2w, const float* __restrict__ c2b,
             const float* __restrict__ s2, const float* __restrict__ bb2,
             const float* __restrict__ m2, const float* __restrict__ v2,
             float* __restrict__ h2out) {
  const int b = blockIdx.x;
  const int k = blockIdx.y;
  const int tid = threadIdx.x;

  __shared__ float patch[3 * 31 * 31];   // 2883
  __shared__ float w1s[16 * 108];        // 1728
  __shared__ float al1[16], be1[16];
  __shared__ float c1o[16 * 13 * 13];    // 2704
  __shared__ float plo[16 * 7 * 7];      // 784
  __shared__ float w2s[16 * 144];        // 2304
  __shared__ float al2[16], be2[16];
  __shared__ float c2o[16 * 5 * 5];      // 400

  // crop start (trunc == floor for the nonneg values here)
  float py = points[b * (2 * KPTS) + 2 * k + 0];
  float px = points[b * (2 * KPTS) + 2 * k + 1];
  int r0 = (int)((py + 0.5f) * (float)HW) - 15;
  int c0 = (int)((px + 0.5f) * (float)HW) - 15;
  r0 = min(max(r0, 0), HW - PATCH);
  c0 = min(max(c0, 0), HW - PATCH);

  for (int i = tid; i < 3 * 31 * 31; i += 128) {
    int c = i / 961, r = (i % 961) / 31, cx = i % 31;
    patch[i] = x[(((size_t)b * 3 + c) * HW + (r0 + r)) * HW + (c0 + cx)];
  }
  for (int i = tid; i < 16 * 108; i += 128) w1s[i] = c1w[k * 16 * 108 + i];
  for (int i = tid; i < 16 * 144; i += 128) w2s[i] = c2w[k * 16 * 144 + i];
  if (tid < 16) {
    float inv = s1[k * 16 + tid] * rsqrtf(v1[k * 16 + tid] + 1e-5f);
    al1[tid] = inv;
    be1[tid] = (c1b[k * 16 + tid] - m1[k * 16 + tid]) * inv + bb1[k * 16 + tid];
    float inv2 = s2[k * 16 + tid] * rsqrtf(v2[k * 16 + tid] + 1e-5f);
    al2[tid] = inv2;
    be2[tid] = (c2b[k * 16 + tid] - m2[k * 16 + tid]) * inv2 + bb2[k * 16 + tid];
  }
  __syncthreads();

  // conv1: 6x6 stride 2 -> [16][13][13], bn+relu
  for (int o = tid; o < 16 * 169; o += 128) {
    int oc = o / 169, r = o % 169, oy = r / 13, ox = r % 13;
    float sum = 0.f;
    const float* w = &w1s[oc * 108];
    for (int ci = 0; ci < 3; ++ci)
      #pragma unroll
      for (int ky = 0; ky < 6; ++ky) {
        const float* pr = &patch[ci * 961 + (2 * oy + ky) * 31 + 2 * ox];
        const float* wr = &w[ci * 36 + ky * 6];
        #pragma unroll
        for (int kx = 0; kx < 6; ++kx) sum += pr[kx] * wr[kx];
      }
    c1o[o] = fmaxf(sum * al1[oc] + be1[oc], 0.f);
  }
  __syncthreads();

  // maxpool 2x2 s2 ceil: 13 -> 7
  for (int o = tid; o < 16 * 49; o += 128) {
    int oc = o / 49, r = o % 49, pyi = r / 7, pxi = r % 7;
    int y0 = 2 * pyi, x0 = 2 * pxi;
    const float* base = &c1o[oc * 169];
    float m = base[y0 * 13 + x0];
    if (x0 + 1 < 13) m = fmaxf(m, base[y0 * 13 + x0 + 1]);
    if (y0 + 1 < 13) {
      m = fmaxf(m, base[(y0 + 1) * 13 + x0]);
      if (x0 + 1 < 13) m = fmaxf(m, base[(y0 + 1) * 13 + x0 + 1]);
    }
    plo[o] = m;
  }
  __syncthreads();

  // conv2: 3x3 -> [16][5][5], bn+relu
  for (int o = tid; o < 16 * 25; o += 128) {
    int oc = o / 25, r = o % 25, oy = r / 5, ox = r % 5;
    float sum = 0.f;
    const float* w = &w2s[oc * 144];
    for (int ci = 0; ci < 16; ++ci)
      #pragma unroll
      for (int ky = 0; ky < 3; ++ky) {
        const float* pr = &plo[ci * 49 + (oy + ky) * 7 + ox];
        const float* wr = &w[ci * 9 + ky * 3];
        sum += pr[0] * wr[0] + pr[1] * wr[1] + pr[2] * wr[2];
      }
    c2o[o] = fmaxf(sum * al2[oc] + be2[oc], 0.f);
  }
  __syncthreads();

  // avgpool 2x2 s2 ceil (clipped-count divisor): 5 -> 3, flatten [16*3*3]
  float* dst = h2out + ((size_t)k * BATCH + b) * 144;
  for (int o = tid; o < 16 * 9; o += 128) {
    int oc = o / 9, r = o % 9, pyi = r / 3, pxi = r % 3;
    int y0 = 2 * pyi, x0 = 2 * pxi;
    int rc = (y0 + 1 < 5) ? 2 : 1, ccnt = (x0 + 1 < 5) ? 2 : 1;
    const float* base = &c2o[oc * 25];
    float s = 0.f;
    for (int yy = 0; yy < rc; ++yy)
      for (int xx = 0; xx < ccnt; ++xx) s += base[(y0 + yy) * 5 + x0 + xx];
    dst[o] = s / (float)(rc * ccnt);
  }
}

// ---------------- WMMA fragment loaders (16x16x32 f16, wave32 layouts) ----------
__device__ __forceinline__ v16h load_a_frag(const _Float16* base, int stride,
                                            int mBase, int kBase, int lane) {
  // A 16x32: lanes 0-15 row M=lane hold K{0-7,16-23}; lanes 16-31 hold K{8-15,24-31}
  const _Float16* p = base + (mBase + (lane & 15)) * stride + kBase + ((lane < 16) ? 0 : 8);
  v16h a;
  #pragma unroll
  for (int e = 0; e < 8; ++e) { a[e] = p[e]; a[e + 8] = p[e + 16]; }
  return a;
}
__device__ __forceinline__ v16h load_b_frag(const _Float16* base, int stride,
                                            int nBase, int kBase, int lane) {
  // B 32x16: column N=lane&15; lanes 0-15 hold K{0-7,16-23}; lanes 16-31 hold K{8-15,24-31}
  const _Float16* p = base + (nBase + (lane & 15)) * stride + kBase + ((lane < 16) ? 0 : 8);
  v16h bf;
  #pragma unroll
  for (int e = 0; e < 8; ++e) { bf[e] = p[e]; bf[e + 8] = p[e + 16]; }
  return bf;
}

// ---------------- Kernel 3: lin1 (144->128) + bn3/relu + lin2 (128->136), WMMA --
// Block = (keypoint k, batch half); 8 waves; 61,440 B static LDS (phase-aliased).
__global__ void __launch_bounds__(256)
lin_kernel(const float* __restrict__ h2, const float* __restrict__ l1w,
           const float* __restrict__ l1b, const float* __restrict__ s3,
           const float* __restrict__ bb3, const float* __restrict__ m3,
           const float* __restrict__ v3, const float* __restrict__ l2w,
           const float* __restrict__ l2b, float* __restrict__ localfea) {
  const int k = blockIdx.x;
  const int hb = blockIdx.y;           // batch half: rows [64*hb, 64*hb+64)
  const int tid = threadIdx.x;
  const int lane = tid & 31;
  const int wv = tid >> 5;

  __shared__ __align__(16) char smem[61440];
  _Float16* Xs  = (_Float16*)smem;              // phase1: [64][160]  (K padded 144->160)
  _Float16* W1s = (_Float16*)(smem + 20480);    // phase1: [128][160]
  _Float16* Y1s = (_Float16*)smem;              // phase2: [64][128]
  _Float16* W2s = (_Float16*)(smem + 16384);    // phase2: [144][128] (N padded 136->144)

  for (int i = tid; i < 64 * 160; i += 256) {
    int r = i / 160, c = i % 160;
    float v = (c < 144) ? h2[((size_t)k * BATCH + hb * 64 + r) * 144 + c] : 0.f;
    Xs[i] = (_Float16)v;
  }
  for (int i = tid; i < 128 * 160; i += 256) {
    int r = i / 160, c = i % 160;
    float v = (c < 144) ? l1w[((size_t)k * 128 + r) * 144 + c] : 0.f;
    W1s[i] = (_Float16)v;
  }
  __syncthreads();

  // lin1: Y1[64,128] = X[64,144] * W1^T ; each wave owns 4 tiles (accums in VGPRs)
  const int mt = wv & 3;
  const int ntb = (wv >> 2) * 4;
  v8f acc[4] = {};
  for (int ks = 0; ks < 5; ++ks) {
    v16h a = load_a_frag(Xs, 160, mt * 16, ks * 32, lane);
    #pragma unroll
    for (int j = 0; j < 4; ++j) {
      v16h bf = load_b_frag(W1s, 160, (ntb + j) * 16, ks * 32, lane);
      acc[j] = __builtin_amdgcn_wmma_f32_16x16x32_f16(false, a, false, bf,
                                                      (short)0, acc[j], false, false);
    }
  }
  __syncthreads();   // all waves done reading Xs/W1s; LDS re-aliased below

  // bias + bn3 + relu -> Y1 (f16) in LDS; stage W2 (f16, row-padded to 144)
  #pragma unroll
  for (int j = 0; j < 4; ++j) {
    int N = (ntb + j) * 16 + (lane & 15);
    float bias = l1b[k * 128 + N];
    float inv  = s3[k * 128 + N] * rsqrtf(v3[k * 128 + N] + 1e-5f);
    float mu   = m3[k * 128 + N];
    float bt   = bb3[k * 128 + N];
    #pragma unroll
    for (int e = 0; e < 8; ++e) {
      int M = mt * 16 + e + ((lane < 16) ? 0 : 8);
      float y = (acc[j][e] + bias - mu) * inv + bt;
      Y1s[M * 128 + N] = (_Float16)fmaxf(y, 0.f);
    }
  }
  for (int i = tid; i < 144 * 128; i += 256) {
    int r = i / 128, c = i % 128;
    float v = (r < 136) ? l2w[((size_t)k * 136 + r) * 128 + c] : 0.f;
    W2s[i] = (_Float16)v;
  }
  __syncthreads();

  // lin2: Y2[64,136] = Y1[64,128] * W2^T ; 36 tiles round-robined over 8 waves
  for (int t = wv; t < 4 * 9; t += 8) {
    int mt2 = t / 9, nt2 = t % 9;
    v8f a2 = {};
    for (int ks = 0; ks < 4; ++ks) {
      v16h a  = load_a_frag(Y1s, 128, mt2 * 16, ks * 32, lane);
      v16h bf = load_b_frag(W2s, 128, nt2 * 16, ks * 32, lane);
      a2 = __builtin_amdgcn_wmma_f32_16x16x32_f16(false, a, false, bf,
                                                  (short)0, a2, false, false);
    }
    int N = nt2 * 16 + (lane & 15);
    if (N < 136) {
      float bias = l2b[k * 136 + N];
      #pragma unroll
      for (int e = 0; e < 8; ++e) {
        int M = mt2 * 16 + e + ((lane < 16) ? 0 : 8);
        int bidx = hb * 64 + M;
        localfea[((size_t)k * BATCH + bidx) * 136 + N] = a2[e] + bias;
      }
    }
  }
}

// ---------------- Kernel 4: concat [2208||136] -> per-keypoint 2344->2 ---------
__global__ void __launch_bounds__(256)
final_kernel(const float* __restrict__ outmean, const float* __restrict__ localfea,
             const float* __restrict__ pw, const float* __restrict__ pb,
             float* __restrict__ out) {
  const int tid = threadIdx.x;
  const int wid = blockIdx.x * 8 + (tid >> 5);
  const int lane = tid & 31;
  if (wid >= BATCH * KPTS) return;                 // grid exact; never taken
  const int b = wid / KPTS, k = wid % KPTS;
  const float* w0 = pw + ((size_t)k * 2 + 0) * 2344;
  const float* w1 = pw + ((size_t)k * 2 + 1) * 2344;
  const float* om = outmean + (size_t)b * 2208;
  const float* lf = localfea + ((size_t)k * BATCH + b) * 136;
  float a0 = 0.f, a1 = 0.f;
  for (int i = lane; i < 2344; i += 32) {
    float f = (i < 2208) ? om[i] : lf[i - 2208];
    a0 += f * w0[i];
    a1 += f * w1[i];
  }
  #pragma unroll
  for (int off = 16; off > 0; off >>= 1) {
    a0 += __shfl_xor(a0, off, 32);
    a1 += __shfl_xor(a1, off, 32);
  }
  if (lane == 0) {
    out[(size_t)b * (2 * KPTS) + 2 * k + 0] = a0 + pb[k * 2 + 0];
    out[(size_t)b * (2 * KPTS) + 2 * k + 1] = a1 + pb[k * 2 + 1];
  }
}

extern "C" void kernel_launch(void* const* d_in, const int* in_sizes, int n_in,
                              void* d_out, int out_size, void* d_ws, size_t ws_size,
                              hipStream_t stream) {
  (void)in_sizes; (void)n_in; (void)out_size; (void)ws_size;
  const float* x        = (const float*)d_in[0];
  const float* features = (const float*)d_in[1];
  const float* points   = (const float*)d_in[2];
  const float* c1w = (const float*)d_in[3];
  const float* c1b = (const float*)d_in[4];
  const float* s1  = (const float*)d_in[5];
  const float* bb1 = (const float*)d_in[6];
  const float* m1  = (const float*)d_in[7];
  const float* v1  = (const float*)d_in[8];
  const float* c2w = (const float*)d_in[9];
  const float* c2b = (const float*)d_in[10];
  const float* s2  = (const float*)d_in[11];
  const float* bb2 = (const float*)d_in[12];
  const float* m2  = (const float*)d_in[13];
  const float* v2  = (const float*)d_in[14];
  const float* l1w = (const float*)d_in[15];
  const float* l1b = (const float*)d_in[16];
  const float* s3  = (const float*)d_in[17];
  const float* bb3 = (const float*)d_in[18];
  const float* m3  = (const float*)d_in[19];
  const float* v3  = (const float*)d_in[20];
  const float* l2w = (const float*)d_in[21];
  const float* l2b = (const float*)d_in[22];
  const float* pw  = (const float*)d_in[23];
  const float* pb  = (const float*)d_in[24];
  float* out = (float*)d_out;

  char* ws = (char*)d_ws;
  float* outmean  = (float*)ws;                                           // B*2208 f32
  float* h2       = (float*)(ws + (size_t)BATCH * 2208 * 4);              // K*B*144 f32
  float* localfea = (float*)(ws + (size_t)BATCH * 2208 * 4
                                + (size_t)KPTS * BATCH * 144 * 4);        // K*B*136 f32

  fmean_kernel<<<dim3((BATCH * 2208) / 32), dim3(256), 0, stream>>>(features, outmean);
  pconv_kernel<<<dim3(BATCH, KPTS), dim3(128), 0, stream>>>(
      x, points, c1w, c1b, s1, bb1, m1, v1, c2w, c2b, s2, bb2, m2, v2, h2);
  lin_kernel<<<dim3(KPTS, 2), dim3(256), 0, stream>>>(
      h2, l1w, l1b, s3, bb3, m3, v3, l2w, l2b, localfea);
  final_kernel<<<dim3(256), dim3(256), 0, stream>>>(outmean, localfea, pw, pb, out);
}
